// LearnableEdgeDetectionSuperpixelTokenizer_21277267984858
// MI455X (gfx1250) — matrix operand order repaired and checked by reading, not compiled, not run on hardware
//
#include <hip/hip_runtime.h>
#include <hip/hip_bf16.h>
#include <utility>

typedef __attribute__((ext_vector_type(16))) _Float16 v16h;
typedef __attribute__((ext_vector_type(8)))  _Float16 v8h;
typedef __attribute__((ext_vector_type(8)))  float    v8f;

#define HH 224
#define WW 224
#define HWP (HH*WW)           // 50176
#define KSEG 196
#define NBR 10
#define PADW 228
#define PLANE (PADW*PADW)     // 51984
#define C2IN 64
#define C2OUT 768
#define KDIM 576              // 9 taps * 64 ch, tap-major
#define KCHUNKS 18            // 576/32
#define MTILES 48             // 768/16
#define EMBD 768
#define EMB2 1536
#define BKROWS 392            // 2*196

__device__ __forceinline__ float bnsc(float g){ return g * rsqrtf(1.0f + 1e-5f); }

__device__ __forceinline__ v8f wmma_f16(v16h a, v16h b, v8f c){
  return __builtin_amdgcn_wmma_f32_16x16x32_f16(false, a, false, b, (short)0, c, false, false);
}

// ---------------- generic fills ----------------
__global__ void k_fill_i32(int* p, int v, int n){
  int i = blockIdx.x*blockDim.x + threadIdx.x; if (i < n) p[i] = v;
}
__global__ void k_fill_f32(float* p, float v, int n){
  int i = blockIdx.x*blockDim.x + threadIdx.x; if (i < n) p[i] = v;
}

// ---------------- grayscale ----------------
__global__ void k_gray(const float* img, float* gray){
  int i = blockIdx.x*blockDim.x + threadIdx.x; if (i >= 2*HWP) return;
  int b = i / HWP, p = i % HWP;
  const float* ip = img + (size_t)b*3*HWP + p;
  gray[i] = (ip[0] + ip[HWP] + ip[2*HWP]) * (1.0f/3.0f);
}

// ---------------- edge CNN ----------------
__global__ void k_econv0(const float* gray, const float* w, const float* g, const float* be, float* out){
  int i = blockIdx.x*blockDim.x + threadIdx.x; if (i >= 2*32*HWP) return;
  int p = i % HWP, oc = (i/HWP) % 32, b = i/(32*HWP);
  int y = p / WW, x = p % WW;
  const float* gp = gray + (size_t)b*HWP;
  float acc = 0.f;
  #pragma unroll
  for (int ky=0; ky<3; ++ky){
    int iy = y + ky - 1; if (iy < 0 || iy >= HH) continue;
    #pragma unroll
    for (int kx=0; kx<3; ++kx){
      int ix = x + kx - 1; if (ix < 0 || ix >= WW) continue;
      acc += w[(oc*3+ky)*3+kx] * gp[iy*WW+ix];
    }
  }
  float v = acc * bnsc(g[oc]) + be[oc];
  out[i] = fmaxf(v, 0.f);
}

__global__ void k_edw(const float* in, const float* w, const float* g, const float* be, float* out){
  int i = blockIdx.x*blockDim.x + threadIdx.x; if (i >= 2*32*HWP) return;
  int p = i % HWP, c = (i/HWP) % 32, b = i/(32*HWP);
  int y = p / WW, x = p % WW;
  const float* ip = in + ((size_t)b*32 + c)*HWP;
  float acc = 0.f;
  #pragma unroll
  for (int ky=0; ky<3; ++ky){
    int iy = y + ky - 1; if (iy < 0 || iy >= HH) continue;
    #pragma unroll
    for (int kx=0; kx<3; ++kx){
      int ix = x + kx - 1; if (ix < 0 || ix >= WW) continue;
      acc += w[(c*3+ky)*3+kx] * ip[iy*WW+ix];
    }
  }
  float v = acc * bnsc(g[c]) + be[c];
  out[i] = fmaxf(v, 0.f);
}

__global__ void k_epw(const float* in, const float* w, const float* g, const float* be, float* out){
  int i = blockIdx.x*blockDim.x + threadIdx.x; if (i >= 2*32*HWP) return;
  int p = i % HWP, oc = (i/HWP) % 32, b = i/(32*HWP);
  const float* ip = in + (size_t)b*32*HWP + p;
  float acc = 0.f;
  #pragma unroll
  for (int ic=0; ic<32; ++ic) acc += w[oc*32+ic] * ip[(size_t)ic*HWP];
  float v = acc * bnsc(g[oc]) + be[oc];
  out[i] = fmaxf(v, 0.f);
}

__global__ void k_pred(const float* in, const float* w, const float* pb,
                       float* grad, float* wgt){
  int i = blockIdx.x*blockDim.x + threadIdx.x; if (i >= 2*HWP) return;
  int b = i / HWP, p = i % HWP;
  const float* ip = in + (size_t)b*32*HWP + p;
  float acc = pb[0];
  #pragma unroll
  for (int c=0; c<32; ++c) acc += w[c] * ip[(size_t)c*HWP];
  float gsig = 1.0f / (1.0f + __expf(-acc));
  grad[i] = gsig;
  float g2 = gsig * gsig;
  wgt[i] = g2 * g2 * 10.0f;   // edge_exponent=4, gradient_weight=10
}

// ---------------- centroid relocation (sequential per image) ----------------
__global__ void k_minima(const float* grad, int* occ, int* cents){
  int b = blockIdx.x;
  const float* g = grad + (size_t)b*HWP;
  int* oc = occ + (size_t)b*HWP;
  int tid = threadIdx.x;
  __shared__ float sred[256];
  __shared__ int   sidx[256];
  __shared__ float s_mn;
  for (int k=0; k<KSEG; ++k){
    int y = 16*(k/14) + 8, x = 16*(k%14) + 8;
    int y0 = max(0, y-NBR), y1 = min(HH, y+NBR);
    int x0 = max(0, x-NBR), x1 = min(WW, x+NBR);
    int ww = x1 - x0, n = ww * (y1 - y0);
    float mn = INFINITY;
    for (int t = tid; t < n; t += 256){
      int yy = y0 + t/ww, xx = x0 + t%ww;
      mn = fminf(mn, g[yy*WW+xx]);
    }
    sred[tid] = mn; __syncthreads();
    for (int s = 128; s > 0; s >>= 1){
      if (tid < s) sred[tid] = fminf(sred[tid], sred[tid+s]);
      __syncthreads();
    }
    if (tid == 0) s_mn = sred[0];
    __syncthreads();
    float gmn = s_mn;
    int best = 0x7fffffff;
    for (int t = tid; t < n; t += 256){
      int yy = y0 + t/ww, xx = x0 + t%ww;
      int idx = yy*WW + xx;
      if (g[idx] == gmn && oc[idx] == 0) best = min(best, idx);
    }
    sidx[tid] = best; __syncthreads();
    for (int s = 128; s > 0; s >>= 1){
      if (tid < s) sidx[tid] = min(sidx[tid], sidx[tid+s]);
      __syncthreads();
    }
    if (tid == 0){
      int idx = sidx[0];
      int ny = y, nx = x;
      if (idx != 0x7fffffff){ ny = idx / WW; nx = idx % WW; oc[idx] = 1; }
      cents[((size_t)b*KSEG + k)*2 + 0] = ny;
      cents[((size_t)b*KSEG + k)*2 + 1] = nx;
    }
    __syncthreads();
  }
}

// ---------------- Voronoi propagation ----------------
__global__ void k_seed(const int* cents, float* dist, int* mask){
  int b = blockIdx.x;
  if (threadIdx.x == 0){
    for (int k=0; k<KSEG; ++k){
      int y = min(max(cents[((size_t)b*KSEG+k)*2+0], 0), HH-1);
      int x = min(max(cents[((size_t)b*KSEG+k)*2+1], 0), WW-1);
      size_t idx = (size_t)b*HWP + y*WW + x;
      dist[idx] = 0.f; mask[idx] = k;
    }
  }
}

__global__ void k_prop(const float* din, const int* min_, float* dout, int* mout,
                       const float* wgt, int dy, int dx){
  int i = blockIdx.x*blockDim.x + threadIdx.x; if (i >= 2*HWP) return;
  int b = i / HWP, p = i % HWP, y = p / WW, x = p % WW;
  int sy = (y - dy + HH) % HH, sx = (x - dx + WW) % WW;   // jnp.roll wrap
  size_t si = (size_t)b*HWP + sy*WW + sx;
  float sd = din[si]; int sm = min_[si];
  float d = din[i];   int m  = min_[i];
  float wd = sd + wgt[i];
  if (sm != -1 && wd < d){ d = wd; m = sm; }
  dout[i] = d; mout[i] = m;
}

__global__ void k_maskfin(const int* m, int* mf){
  int i = blockIdx.x*blockDim.x + threadIdx.x; if (i >= 2*HWP) return;
  int v = m[i]; mf[i] = (v == -1) ? 0 : v;
}

__global__ void k_counts(const int* maskf, const float* grad, float* counts, float* simsum){
  int i = blockIdx.x*blockDim.x + threadIdx.x; if (i >= 2*HWP) return;
  int b = i / HWP;
  int seg = maskf[i];
  int row = b*KSEG + seg;
  atomicAdd(&counts[row], 1.0f);
  float s = fminf(fmaxf(1.0f - grad[i], 0.0f), 1.0f);
  atomicAdd(&simsum[row], s);
}

// ------- conv1 (3->64, 3x3 pad1) -> padded NHWC f16 [b][228][228][64] ------
__global__ void k_conv1(const float* img, const float* w1, const float* b1,
                        const float* g1, const float* be1, _Float16* nhwc){
  int i = blockIdx.x*blockDim.x + threadIdx.x; if (i >= 2*C2IN*HWP) return;
  int p = i % HWP, oc = (i/HWP) % C2IN, b = i/(C2IN*HWP);
  int y = p / WW, x = p % WW;
  float acc = b1[oc];
  #pragma unroll
  for (int ic=0; ic<3; ++ic){
    const float* ip = img + ((size_t)b*3 + ic)*HWP;
    const float* wp = w1 + ((size_t)oc*3 + ic)*9;
    #pragma unroll
    for (int ky=0; ky<3; ++ky){
      int iy = y + ky - 1; if (iy < 0 || iy >= HH) continue;
      #pragma unroll
      for (int kx=0; kx<3; ++kx){
        int ix = x + kx - 1; if (ix < 0 || ix >= WW) continue;
        acc += wp[ky*3+kx] * ip[iy*WW+ix];
      }
    }
  }
  float v = fmaxf(acc * bnsc(g1[oc]) + be1[oc], 0.f);
  nhwc[(((size_t)b*PADW + (y+2))*PADW + (x+2))*C2IN + oc] = (_Float16)v;
}

// ---------------- pack conv2 weights into A-fragment layout ----------------
// k ordering is tap-major: k = tap*64 + c ; A lanes hold k = 16*(lane/16) + element (contiguous).
__global__ void k_packA(const float* w2, _Float16* Apack){
  int i = blockIdx.x*blockDim.x + threadIdx.x;
  if (i >= MTILES*KCHUNKS*32*16) return;
  int h = i & 15, lane = (i >> 4) & 31, kc = (i >> 9) % KCHUNKS, mt = i / (16*32*KCHUNKS);
  int m = mt*16 + (lane & 15);
  int k = kc*32 + h + ((lane & 16) ? 16 : 0);
  int tap = k / 64, c = k % 64;
  float v = w2[(((size_t)m*C2IN + c)*3 + tap/3)*3 + (tap%3)];
  Apack[i] = (_Float16)v;
}

__global__ void k_f16cast(const float* src, _Float16* dst, int n){
  int i = blockIdx.x*blockDim.x + threadIdx.x; if (i < n) dst[i] = (_Float16)src[i];
}

// ------- conv2 WMMA implicit GEMM (16Mx32N per wave) + fused pooling -------
__global__ void k_conv2_wmma(const _Float16* __restrict__ Apack,
                             const _Float16* __restrict__ nhwc,
                             const float* __restrict__ b2, const float* __restrict__ g2,
                             const float* __restrict__ be2,
                             const int* __restrict__ maskf,
                             float* __restrict__ meansum, int* __restrict__ maxval){
  const int lane = threadIdx.x & 31;
  const int wave = threadIdx.x >> 5;
  const int b  = blockIdx.z;
  const int mt = blockIdx.y;
  const int m0 = mt * 16;
  const int n0 = blockIdx.x * 256 + wave * 32;      // 32 pixels per wave
  const int pix0 = n0 + (lane & 15);
  const int pix1 = pix0 + 16;
  const int y0 = pix0 / WW, x0 = pix0 % WW;
  const int y1 = pix1 / WW, x1 = pix1 % WW;
  const _Float16* base = nhwc + (size_t)b * PLANE * C2IN;
  const _Float16* apackt = Apack + (size_t)mt * KCHUNKS * 512 + lane * 16;

  v8f acc0 = {}, acc1 = {};
  #pragma unroll 1
  for (int kc = 0; kc < KCHUNKS; ++kc){
    v16h a = *(const v16h*)(apackt + (size_t)kc * 512);
    int kb  = kc*32 + ((lane & 16) ? 16 : 0);
    int tap = kb >> 6, c0 = kb & 63;
    int oy = (tap/3)*2, ox = (tap%3)*2;
    const _Float16* bp0 = base + ((size_t)((y0+oy)*PADW + (x0+ox)))*C2IN + c0;
    const _Float16* bp1 = base + ((size_t)((y1+oy)*PADW + (x1+ox)))*C2IN + c0;
    v16h bf0 = *(const v16h*)bp0;                  // 16 consecutive channels
    v16h bf1 = *(const v16h*)bp1;
    if (kc + 1 < KCHUNKS){                         // gfx1250 global_prefetch_b8
      int kbn = (kc+1)*32 + ((lane & 16) ? 16 : 0);
      int tn = kbn >> 6, cn = kbn & 63;
      __builtin_prefetch(base + ((size_t)((y0+(tn/3)*2)*PADW + (x0+(tn%3)*2)))*C2IN + cn, 0, 1);
      __builtin_prefetch(apackt + (size_t)(kc+1) * 512, 0, 1);
    }
    acc0 = wmma_f16(a, bf0, acc0);
    acc1 = wmma_f16(a, bf1, acc1);
  }

  const float rsq = rsqrtf(1.0f + 1e-5f);
  int seg0 = maskf[(size_t)b*HWP + pix0];
  int seg1 = maskf[(size_t)b*HWP + pix1];
  size_t row0 = (size_t)(b*KSEG + seg0) * EMBD;
  size_t row1 = (size_t)(b*KSEG + seg1) * EMBD;
  #pragma unroll
  for (int v=0; v<8; ++v){
    int ch = m0 + v + ((lane & 16) ? 8 : 0);
    float sc = g2[ch]*rsq, bb = be2[ch], bs = b2[ch];
    float v0 = fmaxf((acc0[v] + bs) * sc + bb, 0.f);
    float v1 = fmaxf((acc1[v] + bs) * sc + bb, 0.f);
    atomicAdd(&meansum[row0 + ch], v0);
    atomicMax(&maxval[row0 + ch], __float_as_int(v0));  // val>=0: int order == float order
    atomicAdd(&meansum[row1 + ch], v1);
    atomicMax(&maxval[row1 + ch], __float_as_int(v1));
  }
}

// ---------------- emb pack (mean|max) -> f16 ----------------
__global__ void k_emb(const float* meansum, const int* maxval, const float* counts,
                      _Float16* emb){
  int i = blockIdx.x*blockDim.x + threadIdx.x; if (i >= BKROWS*EMB2) return;
  int row = i / EMB2, c = i % EMB2;
  float v;
  if (c < EMBD) v = meansum[(size_t)row*EMBD + c] / fmaxf(counts[row], 1e-6f);
  else          v = __int_as_float(maxval[(size_t)row*EMBD + (c - EMBD)]);
  emb[i] = (_Float16)v;
}

// ---------------- fusion GEMM (392x768, K=1536) WMMA + epilogue ------------
__global__ void k_fuse_wmma(const _Float16* __restrict__ emb,
                            const _Float16* __restrict__ fusw,
                            const float* __restrict__ fusb,
                            const float* __restrict__ posw, const float* __restrict__ posb,
                            const float* __restrict__ counts, const float* __restrict__ simsum,
                            const int* __restrict__ cents,
                            float* __restrict__ out){
  const int lane = threadIdx.x & 31;
  const int wave = threadIdx.x >> 5;
  const int m0 = blockIdx.y * 16;
  const int n0 = blockIdx.x * 128 + wave * 16;

  int arow = m0 + (lane & 15);
  bool avalid = arow < BKROWS;
  const _Float16* abase = emb + (size_t)(avalid ? arow : 0) * EMB2;
  int dcol = n0 + (lane & 15);
  const _Float16* bbase = fusw + (size_t)dcol * EMB2 + ((lane & 16) ? 16 : 0);
  int aoff = (lane & 16) ? 16 : 0;
  v8h zz = {};

  v8f acc = {};
  #pragma unroll 1
  for (int kc = 0; kc < EMB2/32; ++kc){
    const _Float16* ap = abase + kc*32 + aoff;
    v8h lo = avalid ? *(const v8h*)(ap)      : zz;
    v8h hi = avalid ? *(const v8h*)(ap + 8)  : zz;
    v16h a = __builtin_shufflevector(lo, hi, 0,1,2,3,4,5,6,7,8,9,10,11,12,13,14,15);
    v16h bf = *(const v16h*)(bbase + kc*32);
    acc = wmma_f16(a, bf, acc);
  }

  #pragma unroll
  for (int v=0; v<8; ++v){
    int row = m0 + v + ((lane & 16) ? 8 : 0);
    if (row >= BKROWS) continue;
    int d = n0 + (lane & 15);
    float Wk = simsum[row] / fmaxf(counts[row], 1.0f);
    float cy = (float)cents[row*2+0] * (1.0f/(float)WW);
    float cx = (float)cents[row*2+1] * (1.0f/(float)HH);
    float pos = posw[d*2+0]*cy + posw[d*2+1]*cx + posb[d];
    out[(size_t)row*EMBD + d] = (acc[v] + fusb[d]) * Wk + pos;
  }
}

// =====================================================================
static inline void* bump(char*& p, size_t bytes){
  void* r = (void*)p;
  p += (bytes + 255) & ~(size_t)255;
  return r;
}
static inline int nb(long n){ return (int)((n + 255) / 256); }

extern "C" void kernel_launch(void* const* d_in, const int* in_sizes, int n_in,
                              void* d_out, int out_size, void* d_ws, size_t ws_size,
                              hipStream_t stream) {
  (void)in_sizes; (void)n_in; (void)out_size; (void)ws_size;
  const float* img    = (const float*)d_in[0];
  const float* cnn_w1 = (const float*)d_in[1];
  const float* cnn_b1 = (const float*)d_in[2];
  const float* cnn_g1 = (const float*)d_in[3];
  const float* cnn_be1= (const float*)d_in[4];
  const float* cnn_w2 = (const float*)d_in[5];
  const float* cnn_b2 = (const float*)d_in[6];
  const float* cnn_g2 = (const float*)d_in[7];
  const float* cnn_be2= (const float*)d_in[8];
  const float* pd_w0  = (const float*)d_in[9];
  const float* pd_g0  = (const float*)d_in[10];
  const float* pd_be0 = (const float*)d_in[11];
  const float* pd_w1d = (const float*)d_in[12];
  const float* pd_g1a = (const float*)d_in[13];
  const float* pd_be1a= (const float*)d_in[14];
  const float* pd_w1p = (const float*)d_in[15];
  const float* pd_g1b = (const float*)d_in[16];
  const float* pd_be1b= (const float*)d_in[17];
  const float* pd_w2d = (const float*)d_in[18];
  const float* pd_g2a = (const float*)d_in[19];
  const float* pd_be2a= (const float*)d_in[20];
  const float* pd_w2p = (const float*)d_in[21];
  const float* pd_g2b = (const float*)d_in[22];
  const float* pd_be2b= (const float*)d_in[23];
  const float* pred_w = (const float*)d_in[24];
  const float* pred_b = (const float*)d_in[25];
  const float* fus_w  = (const float*)d_in[26];
  const float* fus_b  = (const float*)d_in[27];
  const float* pos_w  = (const float*)d_in[28];
  const float* pos_b  = (const float*)d_in[29];
  float* out = (float*)d_out;

  char* wp = (char*)d_ws;
  _Float16* nhwc  = (_Float16*)bump(wp, (size_t)2*PLANE*C2IN*2);
  _Float16* Apack = (_Float16*)bump(wp, (size_t)MTILES*KCHUNKS*32*16*2);
  _Float16* fuswh = (_Float16*)bump(wp, (size_t)EMBD*EMB2*2);
  _Float16* embh  = (_Float16*)bump(wp, (size_t)BKROWS*EMB2*2);
  float* gray   = (float*)bump(wp, (size_t)2*HWP*4);
  float* eA     = (float*)bump(wp, (size_t)2*32*HWP*4);
  float* eB     = (float*)bump(wp, (size_t)2*32*HWP*4);
  float* grad   = (float*)bump(wp, (size_t)2*HWP*4);
  float* wgt    = (float*)bump(wp, (size_t)2*HWP*4);
  float* distA  = (float*)bump(wp, (size_t)2*HWP*4);
  float* distB  = (float*)bump(wp, (size_t)2*HWP*4);
  int*   maskA  = (int*)  bump(wp, (size_t)2*HWP*4);
  int*   maskB  = (int*)  bump(wp, (size_t)2*HWP*4);
  int*   maskF  = (int*)  bump(wp, (size_t)2*HWP*4);
  int*   occ    = (int*)  bump(wp, (size_t)2*HWP*4);
  int*   cents  = (int*)  bump(wp, (size_t)2*KSEG*2*4);
  float* counts = (float*)bump(wp, (size_t)2*KSEG*4);
  float* simsum = (float*)bump(wp, (size_t)2*KSEG*4);
  float* meansum= (float*)bump(wp, (size_t)2*KSEG*EMBD*4);
  int*   maxval = (int*)  bump(wp, (size_t)2*KSEG*EMBD*4);

  // ---- zero / init state (must be re-done every call: deterministic) ----
  k_fill_i32<<<nb((long)2*PLANE*C2IN*2/4),256,0,stream>>>((int*)nhwc, 0, 2*PLANE*C2IN*2/4);
  k_fill_i32<<<nb(2*HWP),256,0,stream>>>(occ, 0, 2*HWP);
  k_fill_f32<<<nb(2*KSEG),256,0,stream>>>(counts, 0.f, 2*KSEG);
  k_fill_f32<<<nb(2*KSEG),256,0,stream>>>(simsum, 0.f, 2*KSEG);
  k_fill_f32<<<nb((long)2*KSEG*EMBD),256,0,stream>>>(meansum, 0.f, 2*KSEG*EMBD);
  k_fill_i32<<<nb((long)2*KSEG*EMBD),256,0,stream>>>(maxval, 0, 2*KSEG*EMBD);
  k_fill_f32<<<nb(2*HWP),256,0,stream>>>(distA, INFINITY, 2*HWP);
  k_fill_i32<<<nb(2*HWP),256,0,stream>>>(maskA, -1, 2*HWP);

  // ---- weight packing ----
  k_packA<<<nb((long)MTILES*KCHUNKS*32*16),256,0,stream>>>(cnn_w2, Apack);
  k_f16cast<<<nb((long)EMBD*EMB2),256,0,stream>>>(fus_w, fuswh, EMBD*EMB2);

  // ---- edge CNN path (drives segmentation) ----
  k_gray<<<nb(2*HWP),256,0,stream>>>(img, gray);
  k_econv0<<<nb((long)2*32*HWP),256,0,stream>>>(gray, pd_w0, pd_g0, pd_be0, eA);
  k_edw  <<<nb((long)2*32*HWP),256,0,stream>>>(eA, pd_w1d, pd_g1a, pd_be1a, eB);
  k_epw  <<<nb((long)2*32*HWP),256,0,stream>>>(eB, pd_w1p, pd_g1b, pd_be1b, eA);
  k_edw  <<<nb((long)2*32*HWP),256,0,stream>>>(eA, pd_w2d, pd_g2a, pd_be2a, eB);
  k_epw  <<<nb((long)2*32*HWP),256,0,stream>>>(eB, pd_w2p, pd_g2b, pd_be2b, eA);
  k_pred <<<nb(2*HWP),256,0,stream>>>(eA, pred_w, pred_b, grad, wgt);

  // ---- centroids + Voronoi flood fill (200 sequential direction passes) ----
  k_minima<<<2,256,0,stream>>>(grad, occ, cents);
  k_seed<<<2,1,0,stream>>>(cents, distA, maskA);
  {
    float* dI = distA; float* dO = distB; int* mI = maskA; int* mO = maskB;
    const int dirs[4][2] = {{-1,0},{1,0},{0,-1},{0,1}};
    for (int it=0; it<50; ++it)
      for (int d=0; d<4; ++d){
        k_prop<<<nb(2*HWP),256,0,stream>>>(dI, mI, dO, mO, wgt, dirs[d][0], dirs[d][1]);
        std::swap(dI,dO); std::swap(mI,mO);
      }
    k_maskfin<<<nb(2*HWP),256,0,stream>>>(mI, maskF);
  }
  k_counts<<<nb(2*HWP),256,0,stream>>>(maskF, grad, counts, simsum);

  // ---- features: conv1 (NHWC f16) then WMMA conv2 with fused pooling ----
  k_conv1<<<nb((long)2*C2IN*HWP),256,0,stream>>>(img, cnn_w1, cnn_b1, cnn_g1, cnn_be1, nhwc);
  {
    dim3 grid(HWP/256, MTILES, 2);
    k_conv2_wmma<<<grid,256,0,stream>>>(Apack, nhwc, cnn_b2, cnn_g2, cnn_be2,
                                        maskF, meansum, maxval);
  }

  // ---- tokenize: emb -> fused GEMM -> scaled + positional ----
  k_emb<<<nb((long)BKROWS*EMB2),256,0,stream>>>(meansum, maxval, counts, embh);
  {
    dim3 grid(EMBD/128, (BKROWS + 15)/16, 1);
    k_fuse_wmma<<<grid,256,0,stream>>>(embh, fuswh, fus_b, pos_w, pos_b,
                                       counts, simsum, cents, out);
  }
}
